// FGCR_48473000903068
// MI455X (gfx1250) — matrix operand order repaired and checked by reading, not compiled, not run on hardware
//
#include <hip/hip_runtime.h>
#include <hip/hip_bf16.h>

typedef __bf16 bf16t;
typedef __attribute__((ext_vector_type(16))) __bf16 v16bf;
typedef __attribute__((ext_vector_type(8)))  __bf16 v8bf;
typedef __attribute__((ext_vector_type(8)))  float  v8f;

#define HEADSN 8
#define DIMV   512
#define NTOK   2048
#define MTOK   256
#define BATCHN 4
#define RROWS  9220          // 4*2048 + 4*256 + 4
#define KROW0  8192
#define CROW0  9216
#define QKVD   1536
#define MLPD   2048
#define ZBH    32            // BATCHN * HEADSN

// ---------------------------------------------------------------------------
// WMMA GEMM: C[M,N] = A[M,K](bf16) * B, batched over gridDim.z (z = zb*H+zh).
// BT=true : B stored as Bt[N][K] (K contiguous).  BT=false: B stored [K][N].
// MODE 0: Cf = acc                (f32)
// MODE 1: Cb = acc                (bf16)
// MODE 2: Cf = acc + bias[col] + resid[row*ldr+col]
// MODE 3: Cb = gelu(acc + bias[col])   (exact erf gelu, bf16)
// One wave computes a 16x64 tile (4 WMMAs per A-load); 4 waves/block cover
// 4 consecutive M-tiles.  Grid: (N/64, ceil(M/64), Z).  N must be %64.
// A rows >= M are address-clamped (their D rows are discarded by the store
// guard), keeping the inner loop free of EXEC manipulation.
// ---------------------------------------------------------------------------
template<int MODE, bool BT>
__global__ void __launch_bounds__(128) gemm_k(
    const bf16t* __restrict__ A, long long sAb, long long sAh, int lda,
    const bf16t* __restrict__ B, long long sBb, long long sBh, int ldb,
    float* __restrict__ Cf, bf16t* __restrict__ Cb,
    long long sCb, long long sCh, int ldc,
    const float* __restrict__ bias, const float* __restrict__ resid, int ldr,
    int M, int N, int K, int H)
{
  (void)N;
  const int z  = blockIdx.z;
  const int zb = z / H, zh = z % H;
  const bf16t* Ab = A + zb * sAb + zh * sAh;
  const bf16t* Bb = B + zb * sBb + zh * sBh;
  const long long coff = zb * sCb + zh * sCh;

  const int wave = threadIdx.x >> 5;
  const int lane = threadIdx.x & 31;
  const int lid  = lane & 15;
  const int half = lane >> 4;
  const int tm   = blockIdx.y * 4 + wave;   // 16-row M tile of this wave
  if (tm * 16 >= M) return;                 // wave-uniform early out
  const int tnb  = blockIdx.x;              // 64-col N chunk of this block

  int arow = tm * 16 + lid;                 // A row held by this lane
  if (arow > M - 1) arow = M - 1;           // clamp: garbage -> discarded rows
  const bf16t* Arow = Ab + (long long)arow * lda;

  v8f acc[4] = {};
  for (int k0 = 0; k0 < K; k0 += 32) {
    // A 16x32 bf16 layout: lanes 0-15 row=lid, elems 0..15 -> K=k0..k0+15;
    // lanes 16-31 same row, K=k0+16..k0+31.
    const bf16t* ap = Arow + k0 + half * 16;
    v8bf a0 = *(const v8bf*)ap;
    v8bf a1 = *(const v8bf*)(ap + 8);
    v16bf av = __builtin_shufflevector(a0, a1,
        0,1,2,3,4,5,6,7,8,9,10,11,12,13,14,15);
#pragma unroll
    for (int n = 0; n < 4; n++) {
      const int bcol = tnb * 64 + n * 16 + lid;
      // B 32x16 bf16 layout: lanes 0-15 col=lid, elems 0..15 -> K=k0..k0+15,
      // lanes 16-31 -> K=k0+16..k0+31.
      v16bf bv;
      if (BT) {
        const bf16t* bp = Bb + (long long)bcol * ldb + k0 + half * 16;
        v8bf b0 = *(const v8bf*)bp;
        v8bf b1 = *(const v8bf*)(bp + 8);
        bv = __builtin_shufflevector(b0, b1,
            0,1,2,3,4,5,6,7,8,9,10,11,12,13,14,15);
      } else {
        const bf16t* bp = Bb + (long long)(k0 + half * 16) * ldb + bcol;
#pragma unroll
        for (int i = 0; i < 16; i++) bv[i] = bp[(long long)i * ldb];
      }
      acc[n] = __builtin_amdgcn_wmma_f32_16x16x32_bf16(
          false, av, false, bv, (short)0, acc[n], false, false);
    }
  }

  // D layout: VGPR r -> row tm*16 + half*8 + r, col = tile_base + lid
#pragma unroll
  for (int n = 0; n < 4; n++) {
    const int col = tnb * 64 + n * 16 + lid;
#pragma unroll
    for (int r = 0; r < 8; r++) {
      int row = tm * 16 + half * 8 + r;
      if (row >= M) continue;
      float v = acc[n][r];
      long long ci = coff + (long long)row * ldc + col;
      if (MODE == 0) {
        Cf[ci] = v;
      } else if (MODE == 1) {
        Cb[ci] = (bf16t)v;
      } else if (MODE == 2) {
        Cf[ci] = v + bias[col] + resid[(long long)row * ldr + col];
      } else {
        float t = v + bias[col];
        Cb[ci] = (bf16t)(0.5f * t * (1.0f + erff(t * 0.70710678118f)));
      }
    }
  }
}

// ---------------------------------------------------------------------------
// LayerNorm over DIMV=512; one block per row; emits f32 (residual) + bf16.
// ---------------------------------------------------------------------------
__global__ void __launch_bounds__(256) ln_k(
    const float* __restrict__ in, const float* __restrict__ g,
    const float* __restrict__ b, float* __restrict__ outF,
    bf16t* __restrict__ outB)
{
  __shared__ float red[256];
  const int row = blockIdx.x, t = threadIdx.x;
  const long long base = (long long)row * DIMV;
  float x0 = in[base + t], x1 = in[base + t + 256];
  red[t] = x0 + x1; __syncthreads();
  for (int o = 128; o; o >>= 1) { if (t < o) red[t] += red[t + o]; __syncthreads(); }
  float mu = red[0] * (1.0f / DIMV); __syncthreads();
  float d0 = x0 - mu, d1 = x1 - mu;
  red[t] = d0 * d0 + d1 * d1; __syncthreads();
  for (int o = 128; o; o >>= 1) { if (t < o) red[t] += red[t + o]; __syncthreads(); }
  float inv = rsqrtf(red[0] * (1.0f / DIMV) + 1e-5f);
  float y0 = d0 * inv * g[t] + b[t];
  float y1 = d1 * inv * g[t + 256] + b[t + 256];
  outF[base + t] = y0;         outF[base + t + 256] = y1;
  outB[base + t] = (bf16t)y0;  outB[base + t + 256] = (bf16t)y1;
}

// ---------------------------------------------------------------------------
// softmax over m=256 for x-attention rows (z=b*8+h, i in [0,2048)).
// Writes attn*rd (bf16) and softmax(24*logit)*rd transposed into d_out atten.
// ---------------------------------------------------------------------------
__global__ void __launch_bounds__(256) softmax_x_k(
    const float* __restrict__ dots, const float* __restrict__ mask,
    const float* __restrict__ kmask, const float* __restrict__ rd,
    bf16t* __restrict__ attn_bf, float* __restrict__ atten_out)
{
  __shared__ float red[256];
  const int i = blockIdx.x, z = blockIdx.y, t = threadIdx.x;
  const int zb = z >> 3;
  const long long drow = ((long long)z * NTOK + i) * MTOK;
  float logit = dots[drow + t] * 0.125f;
  if (mask[zb * NTOK + i] * kmask[zb * MTOK + t] < 0.5f) logit = -1e9f;
  red[t] = logit; __syncthreads();
  for (int o = 128; o; o >>= 1) { if (t < o) red[t] = fmaxf(red[t], red[t + o]); __syncthreads(); }
  float mx = red[0]; __syncthreads();
  float e1  = expf(logit - mx);
  float e24 = expf(24.0f * (logit - mx));
  red[t] = e1; __syncthreads();
  for (int o = 128; o; o >>= 1) { if (t < o) red[t] += red[t + o]; __syncthreads(); }
  float s1 = red[0]; __syncthreads();
  red[t] = e24; __syncthreads();
  for (int o = 128; o; o >>= 1) { if (t < o) red[t] += red[t + o]; __syncthreads(); }
  float s24 = red[0];
  float r = rd[((long long)zb * MTOK + t) * NTOK + i];
  attn_bf[drow + t] = (bf16t)(e1 / s1 * r);
  atten_out[((long long)z * MTOK + t) * NTOK + i] = e24 / s24 * r;   // (b,h,m,n)
}

// softmax over n=2048 for k-attention rows (z, j in [0,256)); 8 elems/thread.
__global__ void __launch_bounds__(256) softmax_kk(
    const float* __restrict__ kd, const float* __restrict__ mask,
    const float* __restrict__ kmask, const float* __restrict__ rd,
    bf16t* __restrict__ kattn_bf)
{
  __shared__ float red[256];
  const int j = blockIdx.x, z = blockIdx.y, t = threadIdx.x;
  const int zb = z >> 3;
  const long long drow = ((long long)z * MTOK + j) * NTOK;
  const float kmj = kmask[zb * MTOK + j];
  float lg[8], lmax = -3.0e38f;
#pragma unroll
  for (int e = 0; e < 8; e++) {
    int i = t + e * 256;
    float v = kd[drow + i] * 0.125f;
    if (mask[zb * NTOK + i] * kmj < 0.5f) v = -1e9f;
    lg[e] = v; lmax = fmaxf(lmax, v);
  }
  red[t] = lmax; __syncthreads();
  for (int o = 128; o; o >>= 1) { if (t < o) red[t] = fmaxf(red[t], red[t + o]); __syncthreads(); }
  float mx = red[0]; __syncthreads();
  float ls = 0.0f;
#pragma unroll
  for (int e = 0; e < 8; e++) { lg[e] = expf(lg[e] - mx); ls += lg[e]; }
  red[t] = ls; __syncthreads();
  for (int o = 128; o; o >>= 1) { if (t < o) red[t] += red[t + o]; __syncthreads(); }
  float s = red[0];
#pragma unroll
  for (int e = 0; e < 8; e++) {
    int i = t + e * 256;
    float p = lg[e] / s * rd[((long long)zb * MTOK + j) * NTOK + i];
    kattn_bf[drow + i] = (bf16t)p;
  }
}

// softmax over m=256 for the single cls row per (b,h); no rd weighting.
__global__ void __launch_bounds__(256) softmax_c_k(
    const float* __restrict__ cd, const float* __restrict__ mask,
    const float* __restrict__ kmask, bf16t* __restrict__ cattn_bf)
{
  __shared__ float red[256];
  const int z = blockIdx.x, t = threadIdx.x;
  const int zb = z >> 3;
  float logit = cd[z * MTOK + t] * 0.125f;
  if (mask[zb * NTOK] * kmask[zb * MTOK + t] < 0.5f) logit = -1e9f;
  red[t] = logit; __syncthreads();
  for (int o = 128; o; o >>= 1) { if (t < o) red[t] = fmaxf(red[t], red[t + o]); __syncthreads(); }
  float mx = red[0]; __syncthreads();
  float e = expf(logit - mx);
  red[t] = e; __syncthreads();
  for (int o = 128; o; o >>= 1) { if (t < o) red[t] += red[t + o]; __syncthreads(); }
  cattn_bf[z * MTOK + t] = (bf16t)(e / red[0]);
}

// f32[K][N] -> bf16[N][K] (transposed) weight conversion
__global__ void __launch_bounds__(256) transpose_k(
    const float* __restrict__ src, bf16t* __restrict__ dst, int K, int N)
{
  long long idx = (long long)blockIdx.x * 256 + threadIdx.x;
  if (idx >= (long long)K * N) return;
  int n = (int)(idx % N); long long k = idx / N;
  dst[(long long)n * K + k] = (bf16t)src[idx];
}

// k_reps[l] = kx_rows * (kmask >= 0.5)
__global__ void __launch_bounds__(256) kreps_k(
    const float* __restrict__ T, const float* __restrict__ kmask,
    float* __restrict__ out)
{
  long long idx = (long long)blockIdx.x * 256 + threadIdx.x;
  if (idx >= (long long)BATCHN * MTOK * DIMV) return;
  int row = (int)(idx >> 9), c = (int)(idx & 511);
  float keep = (kmask[row] >= 0.5f) ? 1.0f : 0.0f;
  out[idx] = T[(long long)(KROW0 + row) * DIMV + c] * keep;
}

__global__ void __launch_bounds__(256) clst_copy_k(
    const float* __restrict__ T, float* __restrict__ out)
{
  int idx = blockIdx.x * 256 + threadIdx.x;
  if (idx >= BATCHN * DIMV) return;
  out[idx] = T[(long long)(CROW0 + (idx >> 9)) * DIMV + (idx & 511)];
}

static inline int gdiv(int a, int b) { return (a + b - 1) / b; }

extern "C" void kernel_launch(void* const* d_in, const int* in_sizes, int n_in,
                              void* d_out, int out_size, void* d_ws, size_t ws_size,
                              hipStream_t stream) {
  (void)in_sizes; (void)n_in; (void)out_size; (void)ws_size;
  const float* x     = (const float*)d_in[0];
  const float* kx    = (const float*)d_in[1];
  const float* rd    = (const float*)d_in[2];
  const float* clst  = (const float*)d_in[3];
  const float* mask  = (const float*)d_in[4];
  const float* kmask = (const float*)d_in[5];
  const float* ln1_g = (const float*)d_in[6];
  const float* ln1_b = (const float*)d_in[7];
  const float* Wqkv  = (const float*)d_in[8];
  const float* Wout  = (const float*)d_in[9];
  const float* bout  = (const float*)d_in[10];
  const float* ln2_g = (const float*)d_in[11];
  const float* ln2_b = (const float*)d_in[12];
  const float* W1    = (const float*)d_in[13];
  const float* b1    = (const float*)d_in[14];
  const float* W2    = (const float*)d_in[15];
  const float* b2    = (const float*)d_in[16];
  float* out = (float*)d_out;

  // ---- workspace layout (256B aligned slabs) ----
  char* w = (char*)d_ws; size_t off = 0;
  auto alloc = [&](size_t bytes) -> void* {
    void* p = w + off; off += (bytes + 255) & ~(size_t)255; return p;
  };
  float* T      = (float*)alloc((size_t)RROWS * DIMV * 4);        // token stream
  float* Tn     = (float*)alloc((size_t)RROWS * DIMV * 4);        // LN output f32
  bf16t* Tn_bf  = (bf16t*)alloc((size_t)RROWS * DIMV * 2);
  bf16t* qkv_bf = (bf16t*)alloc((size_t)RROWS * QKVD * 2);
  float* dots   = (float*)alloc((size_t)ZBH * NTOK * MTOK * 4);
  bf16t* p_bf   = (bf16t*)alloc((size_t)ZBH * NTOK * MTOK * 2);
  bf16t* aout_bf= (bf16t*)alloc((size_t)RROWS * DIMV * 2);        // merged heads
  float* cd     = (float*)alloc((size_t)ZBH * MTOK * 4);
  bf16t* c_bf   = (bf16t*)alloc((size_t)ZBH * MTOK * 2);
  bf16t* H_bf   = (bf16t*)alloc((size_t)RROWS * MLPD * 2);
  bf16t* WqkvT  = (bf16t*)alloc((size_t)2 * QKVD * DIMV * 2);
  bf16t* WoutT  = (bf16t*)alloc((size_t)2 * DIMV * DIMV * 2);
  bf16t* W1T    = (bf16t*)alloc((size_t)2 * MLPD * DIMV * 2);
  bf16t* W2T    = (bf16t*)alloc((size_t)2 * DIMV * MLPD * 2);

  // ---- build combined token buffer: [x | kx | clst] ----
  (void)hipMemcpyAsync(T, x, (size_t)KROW0 * DIMV * 4,
                       hipMemcpyDeviceToDevice, stream);
  (void)hipMemcpyAsync(T + (size_t)KROW0 * DIMV, kx,
                       (size_t)BATCHN * MTOK * DIMV * 4,
                       hipMemcpyDeviceToDevice, stream);
  (void)hipMemcpyAsync(T + (size_t)CROW0 * DIMV, clst,
                       (size_t)BATCHN * DIMV * 4,
                       hipMemcpyDeviceToDevice, stream);

  // ---- weights -> transposed bf16 ----
  for (int l = 0; l < 2; l++) {
    transpose_k<<<gdiv(DIMV * QKVD, 256), 256, 0, stream>>>(
        Wqkv + (size_t)l * DIMV * QKVD, WqkvT + (size_t)l * QKVD * DIMV, DIMV, QKVD);
    transpose_k<<<gdiv(DIMV * DIMV, 256), 256, 0, stream>>>(
        Wout + (size_t)l * DIMV * DIMV, WoutT + (size_t)l * DIMV * DIMV, DIMV, DIMV);
    transpose_k<<<gdiv(DIMV * MLPD, 256), 256, 0, stream>>>(
        W1 + (size_t)l * DIMV * MLPD, W1T + (size_t)l * MLPD * DIMV, DIMV, MLPD);
    transpose_k<<<gdiv(MLPD * DIMV, 256), 256, 0, stream>>>(
        W2 + (size_t)l * MLPD * DIMV, W2T + (size_t)l * DIMV * MLPD, MLPD, DIMV);
  }

  float* atten_base = out + 2 * (size_t)BATCHN * MTOK * DIMV + BATCHN * DIMV;

  for (int l = 0; l < 2; l++) {
    // LN1 on all 9220 rows
    ln_k<<<RROWS, 256, 0, stream>>>(T, ln1_g + l * DIMV, ln1_b + l * DIMV, Tn, Tn_bf);

    // QKV: [9220,512] x [512,1536] -> bf16
    gemm_k<1, true><<<dim3(QKVD / 64, gdiv(RROWS, 64), 1), 128, 0, stream>>>(
        Tn_bf, 0, 0, DIMV, WqkvT + (size_t)l * QKVD * DIMV, 0, 0, DIMV,
        nullptr, qkv_bf, 0, 0, QKVD, nullptr, nullptr, 0, RROWS, QKVD, DIMV, 1);

    // dots[z,i,j] = q_x . k_k   (M=2048,N=256,K=64, z=b*8+h)
    gemm_k<0, true><<<dim3(MTOK / 64, NTOK / 64, ZBH), 128, 0, stream>>>(
        qkv_bf, (long long)NTOK * QKVD, 64, QKVD,
        qkv_bf + (long long)KROW0 * QKVD + 512, (long long)MTOK * QKVD, 64, QKVD,
        dots, nullptr, (long long)HEADSN * NTOK * MTOK, (long long)NTOK * MTOK, MTOK,
        nullptr, nullptr, 0, NTOK, MTOK, 64, HEADSN);

    // softmax + rd weighting; also emits atten (dots*24 softmax, transposed)
    softmax_x_k<<<dim3(NTOK, ZBH), 256, 0, stream>>>(
        dots, mask, kmask, rd, p_bf,
        atten_base + (long long)l * ZBH * MTOK * NTOK);

    // x_ = attn @ v_k   (M=2048,N=64,K=256) -> bf16 merged-head buffer
    gemm_k<1, false><<<dim3(1, NTOK / 64, ZBH), 128, 0, stream>>>(
        p_bf, (long long)HEADSN * NTOK * MTOK, (long long)NTOK * MTOK, MTOK,
        qkv_bf + (long long)KROW0 * QKVD + 1024, (long long)MTOK * QKVD, 64, QKVD,
        nullptr, aout_bf, (long long)NTOK * DIMV, 64, DIMV,
        nullptr, nullptr, 0, NTOK, 64, MTOK, HEADSN);

    // kd[z,j,i] = q_k . k_x   (M=256,N=2048,K=64)
    gemm_k<0, true><<<dim3(NTOK / 64, MTOK / 64, ZBH), 128, 0, stream>>>(
        qkv_bf + (long long)KROW0 * QKVD, (long long)MTOK * QKVD, 64, QKVD,
        qkv_bf + 512, (long long)NTOK * QKVD, 64, QKVD,
        dots, nullptr, (long long)HEADSN * MTOK * NTOK, (long long)MTOK * NTOK, NTOK,
        nullptr, nullptr, 0, MTOK, NTOK, 64, HEADSN);

    softmax_kk<<<dim3(MTOK, ZBH), 256, 0, stream>>>(dots, mask, kmask, rd, p_bf);

    // kx_ = kattn @ v_x   (M=256,N=64,K=2048) -> bf16
    gemm_k<1, false><<<dim3(1, MTOK / 64, ZBH), 128, 0, stream>>>(
        p_bf, (long long)HEADSN * MTOK * NTOK, (long long)MTOK * NTOK, NTOK,
        qkv_bf + 1024, (long long)NTOK * QKVD, 64, QKVD,
        nullptr, aout_bf + (long long)KROW0 * DIMV, (long long)MTOK * DIMV, 64, DIMV,
        nullptr, nullptr, 0, MTOK, 64, NTOK, HEADSN);

    // cd = q_c . k_k  (M=1,N=256,K=64)
    gemm_k<0, true><<<dim3(MTOK / 64, 1, ZBH), 128, 0, stream>>>(
        qkv_bf + (long long)CROW0 * QKVD, (long long)QKVD, 64, QKVD,
        qkv_bf + (long long)KROW0 * QKVD + 512, (long long)MTOK * QKVD, 64, QKVD,
        cd, nullptr, (long long)HEADSN * MTOK, MTOK, MTOK,
        nullptr, nullptr, 0, 1, MTOK, 64, HEADSN);

    softmax_c_k<<<ZBH, 256, 0, stream>>>(cd, mask, kmask, c_bf);

    // c_ = cattn @ v_k   (M=1,N=64,K=256) -> bf16
    gemm_k<1, false><<<dim3(1, 1, ZBH), 128, 0, stream>>>(
        c_bf, (long long)HEADSN * MTOK, MTOK, MTOK,
        qkv_bf + (long long)KROW0 * QKVD + 1024, (long long)MTOK * QKVD, 64, QKVD,
        nullptr, aout_bf + (long long)CROW0 * DIMV, DIMV, 64, DIMV,
        nullptr, nullptr, 0, 1, 64, MTOK, HEADSN);

    // Wout + bias + residual(LN1 output) -> new token stream (f32)
    gemm_k<2, true><<<dim3(DIMV / 64, gdiv(RROWS, 64), 1), 128, 0, stream>>>(
        aout_bf, 0, 0, DIMV, WoutT + (size_t)l * DIMV * DIMV, 0, 0, DIMV,
        T, nullptr, 0, 0, DIMV, bout + l * DIMV, Tn, DIMV, RROWS, DIMV, DIMV, 1);

    // MLP: LN2 -> W1+gelu (bf16) -> W2 + bias + residual(T)
    ln_k<<<RROWS, 256, 0, stream>>>(T, ln2_g + l * DIMV, ln2_b + l * DIMV, Tn, Tn_bf);
    gemm_k<3, true><<<dim3(MLPD / 64, gdiv(RROWS, 64), 1), 128, 0, stream>>>(
        Tn_bf, 0, 0, DIMV, W1T + (size_t)l * MLPD * DIMV, 0, 0, DIMV,
        nullptr, H_bf, 0, 0, MLPD, b1 + l * MLPD, nullptr, 0, RROWS, MLPD, DIMV, 1);
    gemm_k<2, true><<<dim3(DIMV / 64, gdiv(RROWS, 64), 1), 128, 0, stream>>>(
        H_bf, 0, 0, MLPD, W2T + (size_t)l * DIMV * MLPD, 0, 0, MLPD,
        T, nullptr, 0, 0, DIMV, b2 + l * DIMV, T, DIMV, RROWS, DIMV, MLPD, 1);

    // k_reps[l] = kx * (kmask >= 0.5)
    kreps_k<<<gdiv(BATCHN * MTOK * DIMV, 256), 256, 0, stream>>>(
        T, kmask, out + (long long)l * BATCHN * MTOK * DIMV);
  }

  // final clst
  clst_copy_k<<<gdiv(BATCHN * DIMV, 256), 256, 0, stream>>>(
      T, out + 2 * (size_t)BATCHN * MTOK * DIMV);
}
